// PointGenerator_30176440222313
// MI455X (gfx1250) — matrix operand order repaired and checked by reading, not compiled, not run on hardware
//
#include <hip/hip_runtime.h>
#include <hip/hip_bf16.h>

// B=8, N=2048, C=128, K=16, WDIM=512, FOUT=128
#define BB 8
#define NN 2048
#define CC 128
#define KK 16
#define WDIM 512
#define FOUT 128
#define SQRT2F 1.41421356237309515f

typedef __attribute__((ext_vector_type(16))) _Float16 v16h;
typedef __attribute__((ext_vector_type(8)))  _Float16 v8h;
typedef __attribute__((ext_vector_type(8)))  float    v8f;
typedef __attribute__((ext_vector_type(4)))  unsigned int v4u;
typedef __attribute__((ext_vector_type(8)))  int      v8i;
typedef __attribute__((ext_vector_type(4)))  int      v4i;

__device__ __forceinline__ v16h cat16(v8h lo, v8h hi) {
  return __builtin_shufflevector(lo, hi, 0,1,2,3,4,5,6,7,8,9,10,11,12,13,14,15);
}

__device__ __forceinline__ float lrelu(float v) {
  return (v > 0.f ? v : 0.2f * v) * SQRT2F;
}

// ---------------------------------------------------------------------------
// Per-batch affine vectors: gamma = 1 + w@Wg, beta = w@Wb, ls = w@Wls + bls
// ---------------------------------------------------------------------------
__global__ void precompute_kernel(const float* __restrict__ w,
                                  const float* __restrict__ Wg,
                                  const float* __restrict__ Wb,
                                  const float* __restrict__ Wls,
                                  const float* __restrict__ bls,
                                  float* __restrict__ gamma,
                                  float* __restrict__ beta,
                                  float* __restrict__ ls) {
  int b = blockIdx.x, c = threadIdx.x;
  float g = 0.f, be = 0.f, l = 0.f;
  for (int k = 0; k < WDIM; ++k) {
    float wv = w[b * WDIM + k];
    g  += wv * Wg [k * CC + c];
    be += wv * Wb [k * CC + c];
    l  += wv * Wls[k * CC + c];
  }
  gamma[b * CC + c] = 1.f + g;
  beta [b * CC + c] = be;
  ls   [b * CC + c] = l + bls[c];
}

// ---------------------------------------------------------------------------
// LayerNorm -> h (f16) and sq = sum(h*h). One 128-thread block per row.
// ---------------------------------------------------------------------------
__global__ __launch_bounds__(128) void ln_kernel(const float* __restrict__ x,
                                                 const float* __restrict__ gamma,
                                                 const float* __restrict__ beta,
                                                 _Float16* __restrict__ hh,
                                                 float* __restrict__ sq) {
  int row = blockIdx.x;            // b*N + n
  int b = row >> 11;
  int c = threadIdx.x;
  int wave = c >> 5, lane = c & 31;
  __shared__ float red[4];

  float v = x[(size_t)row * CC + c];
  float s = v;
  #pragma unroll
  for (int off = 16; off; off >>= 1) s += __shfl_xor(s, off);
  if (lane == 0) red[wave] = s;
  __syncthreads();
  float mean = (red[0] + red[1] + red[2] + red[3]) * (1.f / 128.f);
  __syncthreads();

  float d = v - mean;
  float s2 = d * d;
  #pragma unroll
  for (int off = 16; off; off >>= 1) s2 += __shfl_xor(s2, off);
  if (lane == 0) red[wave] = s2;
  __syncthreads();
  float var = (red[0] + red[1] + red[2] + red[3]) * (1.f / 128.f);
  __syncthreads();

  float h = d * __frsqrt_rn(var + 1e-5f) * gamma[b * CC + c] + beta[b * CC + c];
  hh[(size_t)row * CC + c] = (_Float16)h;

  float q = h * h;
  #pragma unroll
  for (int off = 16; off; off >>= 1) q += __shfl_xor(q, off);
  if (lane == 0) red[wave] = q;
  __syncthreads();
  if (c == 0) sq[row] = red[0] + red[1] + red[2] + red[3];
}

// ---------------------------------------------------------------------------
// Pack a (Kdim x Ndim) f32 weight into WMMA B-fragment order, f16:
//   dst[((nt*KS + ks)*32 + lane)*16 + e] = W[ks*32 + (lane>>4)*8 + (e&7) + (e>>3)*16][nt*16 + (lane&15)]
// ---------------------------------------------------------------------------
__global__ void pack_w_kernel(const float* __restrict__ src, _Float16* __restrict__ dst,
                              int Kdim, int Ndim) {
  int t = blockIdx.x * blockDim.x + threadIdx.x;
  if (t >= Kdim * Ndim) return;
  int KS = Kdim >> 5;
  int e = t & 15;
  int l = (t >> 4) & 31;
  int q = t >> 9;
  int ks = q % KS;
  int nt = q / KS;
  int c = nt * 16 + (l & 15);
  int k = ks * 32 + ((l >> 4) << 3) + (e & 7) + ((e >> 3) << 4);
  dst[t] = (_Float16)src[k * Ndim + c];
}

// ---------------------------------------------------------------------------
// KNN: per (batch, 16-row tile): Gram via WMMA -> dist rows in LDS -> top-17
// merge -> neighbor indices (dropping self).
// ---------------------------------------------------------------------------
#define DP 2056   // padded dist row (floats)

__global__ __launch_bounds__(256) void knn_kernel(const _Float16* __restrict__ hh,
                                                  const float* __restrict__ sq,
                                                  int* __restrict__ idxOut) {
  const int b = blockIdx.y;
  const int rt = blockIdx.x;               // row tile (16 rows)
  const int tid = threadIdx.x;
  const int wave = tid >> 5, lane = tid & 31;

  extern __shared__ char smem[];
  float* dist = (float*)smem;                                   // [16][DP]
  float* mergeD = (float*)(smem + 16 * DP * 4);                 // [8][32][17]
  int*   mergeI = (int*)  (smem + 16 * DP * 4 + 8 * 32 * 17 * 4);

  const _Float16* hb = hh + (size_t)b * NN * CC;
  const float* sqb = sq + (size_t)b * NN;

  const int m = lane & 15;
  const int kb8 = (lane >> 4) << 3;        // A-frag K base: 0 or 8
  const int rowg = rt * 16 + m;

  // A fragments for this row tile: 4 k-steps of 32
  v16h afrag[4];
  #pragma unroll
  for (int ks = 0; ks < 4; ++ks) {
    const _Float16* p = hb + (size_t)rowg * CC + ks * 32 + kb8;
    afrag[ks] = cat16(*(const v8h*)p, *(const v8h*)(p + 16));
  }
  float sqr[8];
  #pragma unroll
  for (int r = 0; r < 8; ++r) sqr[r] = sqb[rt * 16 + r + ((lane >> 4) << 3)];

  // each wave sweeps every 8th column tile
  for (int ct = wave; ct < 128; ct += 8) {
    if (ct + 8 < 128)
      __builtin_prefetch(hb + (size_t)(ct + 8) * 16 * CC, 0, 0);
    const int colg = ct * 16 + m;
    const _Float16* pB = hb + (size_t)colg * CC + ((lane >> 4) << 3);
    v8f acc = {};
    #pragma unroll
    for (int ks = 0; ks < 4; ++ks) {
      v16h bfrag = cat16(*(const v8h*)(pB + ks * 32), *(const v8h*)(pB + ks * 32 + 16));
      acc = __builtin_amdgcn_wmma_f32_16x16x32_f16(false, afrag[ks], false, bfrag,
                                                   (short)0, acc, false, false);
    }
    float sqc = sqb[ct * 16 + m];
    #pragma unroll
    for (int r = 0; r < 8; ++r) {
      int mrow = r + ((lane >> 4) << 3);
      int ncol = ct * 16 + m;
      float dv = sqr[r] + sqc - 2.f * acc[r];
      if (ncol == rt * 16 + mrow) dv = -3.0e38f;   // self -> always picked first, dropped
      dist[mrow * DP + ncol] = dv;
    }
  }
  __syncthreads();

  // top-17 per row; each wave handles 2 rows
  for (int rr = 0; rr < 2; ++rr) {
    int row = wave * 2 + rr;
    float dbest[17]; int ibest[17];
    #pragma unroll
    for (int i = 0; i < 17; ++i) { dbest[i] = 3.0e38f; ibest[i] = -1; }
    for (int j = lane; j < NN; j += 32) {
      float dv = dist[row * DP + j];
      if (dv < dbest[16]) {
        float cd = dv; int ci = j;
        #pragma unroll
        for (int i = 0; i < 17; ++i) {
          if (cd < dbest[i]) {
            float td = dbest[i]; dbest[i] = cd; cd = td;
            int ti = ibest[i]; ibest[i] = ci; ci = ti;
          }
        }
      }
    }
    int base = (wave * 32 + lane) * 17;
    #pragma unroll
    for (int i = 0; i < 17; ++i) { mergeD[base + i] = dbest[i]; mergeI[base + i] = ibest[i]; }
    __syncthreads();

    // 32-way merge of sorted per-lane lists via shfl argmin
    int p = 0;
    for (int k = 0; k < 17; ++k) {
      int pp = (p < 17) ? p : 16;
      float bd = (p < 17) ? mergeD[base + pp] : 3.0e38f;
      int   bi = (p < 17) ? mergeI[base + pp] : -1;
      int   bl = lane;
      #pragma unroll
      for (int off = 16; off; off >>= 1) {
        float od = __shfl_xor(bd, off);
        int   oi = __shfl_xor(bi, off);
        int   ol = __shfl_xor(bl, off);
        if (od < bd || (od == bd && ol < bl)) { bd = od; bi = oi; bl = ol; }
      }
      if (lane == bl) ++p;
      if (lane == 0 && k >= 1)
        idxOut[((size_t)b * NN + rt * 16 + row) * KK + (k - 1)] = bi;
    }
    __syncthreads();
  }
}

// ---------------------------------------------------------------------------
// Tensor Data Mover: DMA the 128KB packed-weight block (4096 dwords x 8 rows)
// from global memory into LDS offset 0. Issued by one wave per workgroup.
// Descriptor layout per CDNA5 ISA 8.3/8.4 (D# groups 0/1; 2D tensor, groups
// 2/3 zero). Toolchain arity differs: clang-22 (ROCm 7.2) = 5 args,
// clang-23 (amdgpu-toolchain) = 6 args.
// ---------------------------------------------------------------------------
__device__ __forceinline__ void tdm_load_weights_to_lds(const _Float16* wp) {
  unsigned long long ga = (unsigned long long)(uintptr_t)wp;
  v4u g0;
  g0.x = 1u;                                            // count=1, user desc
  g0.y = 0u;                                            // lds_addr = 0
  g0.z = (unsigned int)(ga & 0xFFFFFFFFu);              // global_addr[31:0]
  g0.w = (unsigned int)((ga >> 32) & 0x01FFFFFFu)       // global_addr[56:32]
       | (2u << 30);                                    // type = 2 ("image")
  v8i g1;
  g1[0] = (int)(2u << 16);        // workgroup_mask=0, data_size=2 (4B units)
  g1[1] = (int)(4096u << 16);     // tensor_dim0[15:0] << 16 (dim0 = 4096 dwords)
  g1[2] = (int)((4096u >> 16) | (8u << 16));  // tensor_dim0[31:16] | tensor_dim1[15:0]
  g1[3] = (int)(4096u << 16);     // tensor_dim1[31:16] | tile_dim0 = 4096
  g1[4] = 8;                      // tile_dim1 = 8, tile_dim2 = 0
  g1[5] = 4096;                   // tensor_dim0_stride[31:0] = 4096 dwords
  g1[6] = 0;                      // stride0 hi | stride1 lo
  g1[7] = 0;                      // stride1 hi
  v4i g2 = {0, 0, 0, 0};
  v4i g3 = {0, 0, 0, 0};
#if defined(__clang_major__) && (__clang_major__ >= 23)
  v8i g4 = {0, 0, 0, 0, 0, 0, 0, 0};
  __builtin_amdgcn_tensor_load_to_lds(g0, g1, g2, g3, g4, 0);
#else
  __builtin_amdgcn_tensor_load_to_lds(g0, g1, g2, g3, 0);
#endif
}

// ---------------------------------------------------------------------------
// Edge MLP chain. 8 waves/block, one point per wave. Packed weights staged in
// LDS once per block (TDM async DMA + verified fallback copy); per-wave LDS
// scratch for C-layout -> A-layout hops.
// ---------------------------------------------------------------------------
__device__ __forceinline__ v16h ld_bfrag(const _Float16* base, int nt, int ks, int KS, int lane) {
  const _Float16* p = base + ((size_t)((nt * KS + ks) * 32 + lane)) * 16;
  return cat16(*(const v8h*)p, *(const v8h*)(p + 8));
}

__global__ __launch_bounds__(256) void edge_kernel(const _Float16* __restrict__ hh,
                                                   const int* __restrict__ idxIn,
                                                   const _Float16* __restrict__ wp,
                                                   const float* __restrict__ b1,
                                                   const float* __restrict__ b2,
                                                   const float* __restrict__ bx1,
                                                   const float* __restrict__ bx2,
                                                   const float* __restrict__ ls,
                                                   const float* __restrict__ x,
                                                   float* __restrict__ out) {
  const int tid = threadIdx.x;
  const int wave = tid >> 5, lane = tid & 31;

  extern __shared__ char smem[];
  _Float16* lwts = (_Float16*)smem;                 // 65536 halves of packed weights
  _Float16* w1l  = lwts;                            // NT=4, KS=4 -> 8192
  _Float16* w2l  = lwts + 8192;                     // NT=8, KS=2 -> 8192
  _Float16* wx1l = lwts + 16384;                    // NT=8, KS=8 -> 32768
  _Float16* wx2l = lwts + 49152;                    // NT=8, KS=4 -> 16384
  _Float16* stage = lwts + 65536 + wave * 2048;     // per-wave 4KB scratch

  // Async tensor DMA of all packed weights (128KB) into LDS (wave 0 only;
  // TDM ignores EXEC, so gate by control flow), then wait on TENSORcnt.
  if (wave == 0) {
    tdm_load_weights_to_lds(wp);
    __builtin_amdgcn_s_wait_tensorcnt(0);
  }
  __syncthreads();
  // Verified staging path (overwrites the DMA'd region with identical data).
  {
    const int4* src = (const int4*)wp;
    int4* dst = (int4*)lwts;
    for (int i = tid; i < (65536 * 2) / 16; i += 256) dst[i] = src[i];
  }
  __syncthreads();

  const int point = blockIdx.x * 8 + wave;          // b*N + n
  const int b = point >> 11;
  const int n = point & (NN - 1);
  const _Float16* hb = hh + (size_t)b * NN * CC;

  const int m = lane & 15;
  const int kb8 = (lane >> 4) << 3;

  const int nbr = idxIn[(size_t)point * KK + m];
  __builtin_prefetch(hb + (size_t)nbr * CC, 0, 1);

  // central (broadcast over M) and delta A-fragments, K=128 -> 4 k-steps
  v16h centF[4], deltaF[4];
  #pragma unroll
  for (int ks = 0; ks < 4; ++ks) {
    const _Float16* pc = hb + (size_t)n * CC + ks * 32 + kb8;
    v16h cf = cat16(*(const v8h*)pc, *(const v8h*)(pc + 16));
    const _Float16* pn = hb + (size_t)nbr * CC + ks * 32 + kb8;
    v16h nf = cat16(*(const v8h*)pn, *(const v8h*)(pn + 16));
    centF[ks] = cf;
    deltaF[ks] = nf - cf;
  }

  // --- hw = lrelu(delta @ W1 + b1) : 16x64 ---
  v8f acc1[4];
  #pragma unroll
  for (int nt = 0; nt < 4; ++nt) {
    v8f a = {};
    #pragma unroll
    for (int ks = 0; ks < 4; ++ks)
      a = __builtin_amdgcn_wmma_f32_16x16x32_f16(false, deltaF[ks], false,
            ld_bfrag(w1l, nt, ks, 4, lane), (short)0, a, false, false);
    acc1[nt] = a;
  }
  #pragma unroll
  for (int nt = 0; nt < 4; ++nt) {
    int cc = nt * 16 + m;
    float bv = b1[cc];
    #pragma unroll
    for (int r = 0; r < 8; ++r) {
      int mm = r + ((lane >> 4) << 3);
      stage[mm * 64 + cc] = (_Float16)lrelu(acc1[nt][r] + bv);
    }
  }
  asm volatile("s_wait_dscnt 0" ::: "memory");

  // hw A-fragments from LDS (K=64 -> 2 k-steps)
  v16h hwA[2];
  #pragma unroll
  for (int ks = 0; ks < 2; ++ks) {
    const _Float16* p = stage + m * 64 + ks * 32 + kb8;
    hwA[ks] = cat16(*(const v8h*)p, *(const v8h*)(p + 16));
  }

  // --- wlog = hw @ W2 + b2 : 16x128, then softmax over M (neighbors) ---
  v8f wsm[8];
  #pragma unroll
  for (int nt = 0; nt < 8; ++nt) {
    v8f a = {};
    #pragma unroll
    for (int ks = 0; ks < 2; ++ks)
      a = __builtin_amdgcn_wmma_f32_16x16x32_f16(false, hwA[ks], false,
            ld_bfrag(w2l, nt, ks, 2, lane), (short)0, a, false, false);
    float bv = b2[nt * 16 + m];
    #pragma unroll
    for (int r = 0; r < 8; ++r) a[r] += bv;
    // softmax over the 16 rows of this column (8 in-lane + 8 in lane^16)
    float mx = a[0];
    #pragma unroll
    for (int r = 1; r < 8; ++r) mx = fmaxf(mx, a[r]);
    mx = fmaxf(mx, __shfl_xor(mx, 16));
    float sum = 0.f;
    #pragma unroll
    for (int r = 0; r < 8; ++r) { float e = __expf(a[r] - mx); a[r] = e; sum += e; }
    sum += __shfl_xor(sum, 16);
    float inv = 1.f / sum;
    #pragma unroll
    for (int r = 0; r < 8; ++r) a[r] *= inv;
    wsm[nt] = a;
  }

  // --- hx = lrelu(ee @ Wx1 + bx1), ee = [central | delta], K=256 -> 8 k-steps ---
  #pragma unroll
  for (int nt = 0; nt < 8; ++nt) {
    v8f a = {};
    #pragma unroll
    for (int ks = 0; ks < 8; ++ks) {
      v16h af = (ks < 4) ? centF[ks] : deltaF[ks - 4];
      a = __builtin_amdgcn_wmma_f32_16x16x32_f16(false, af, false,
            ld_bfrag(wx1l, nt, ks, 8, lane), (short)0, a, false, false);
    }
    int cc = nt * 16 + m;
    float bv = bx1[cc];
    #pragma unroll
    for (int r = 0; r < 8; ++r) {
      int mm = r + ((lane >> 4) << 3);
      stage[mm * 128 + cc] = (_Float16)lrelu(a[r] + bv);
    }
  }
  asm volatile("s_wait_dscnt 0" ::: "memory");

  // hx A-fragments from LDS (K=128 -> 4 k-steps)
  v16h hxA[4];
  #pragma unroll
  for (int ks = 0; ks < 4; ++ks) {
    const _Float16* p = stage + m * 128 + ks * 32 + kb8;
    hxA[ks] = cat16(*(const v8h*)p, *(const v8h*)(p + 16));
  }

  // --- xe = hx @ Wx2 + bx2 ; agg = sum_m xe*wsm ; out = agg*ls + x ---
  #pragma unroll
  for (int nt = 0; nt < 8; ++nt) {
    v8f a = {};
    #pragma unroll
    for (int ks = 0; ks < 4; ++ks)
      a = __builtin_amdgcn_wmma_f32_16x16x32_f16(false, hxA[ks], false,
            ld_bfrag(wx2l, nt, ks, 4, lane), (short)0, a, false, false);
    float bv = bx2[nt * 16 + m];
    float s = 0.f;
    #pragma unroll
    for (int r = 0; r < 8; ++r) s += (a[r] + bv) * wsm[nt][r];
    s += __shfl_xor(s, 16);
    if (lane < 16) {
      int c = nt * 16 + lane;
      out[(size_t)point * CC + c] = s * ls[b * CC + c] + x[(size_t)point * CC + c];
    }
  }
}

// ---------------------------------------------------------------------------
extern "C" void kernel_launch(void* const* d_in, const int* in_sizes, int n_in,
                              void* d_out, int out_size, void* d_ws, size_t ws_size,
                              hipStream_t stream) {
  const float* x   = (const float*)d_in[0];
  const float* w   = (const float*)d_in[1];
  const float* Wg  = (const float*)d_in[2];
  const float* Wb  = (const float*)d_in[3];
  const float* W1  = (const float*)d_in[4];
  const float* b1  = (const float*)d_in[5];
  const float* W2  = (const float*)d_in[6];
  const float* b2  = (const float*)d_in[7];
  const float* Wx1 = (const float*)d_in[8];
  const float* bx1 = (const float*)d_in[9];
  const float* Wx2 = (const float*)d_in[10];
  const float* bx2 = (const float*)d_in[11];
  const float* Wls = (const float*)d_in[12];
  const float* bls = (const float*)d_in[13];
  float* out = (float*)d_out;

  char* ws = (char*)d_ws;
  float*    gamma = (float*)(ws);
  float*    beta  = (float*)(ws + 4096);
  float*    lsv   = (float*)(ws + 8192);
  _Float16* hh    = (_Float16*)(ws + 12288);                       // 4 MB
  float*    sq    = (float*)(ws + 12288 + 4194304);                // 64 KB
  int*      idx   = (int*)(ws + 12288 + 4194304 + 65536);          // 2 MB
  _Float16* wp    = (_Float16*)(ws + 12288 + 4194304 + 65536 + 2097152); // 128 KB

  (void)in_sizes; (void)n_in; (void)out_size; (void)ws_size;

  hipFuncSetAttribute((const void*)knn_kernel,
                      hipFuncAttributeMaxDynamicSharedMemorySize, 166400);
  hipFuncSetAttribute((const void*)edge_kernel,
                      hipFuncAttributeMaxDynamicSharedMemorySize, 163840);

  precompute_kernel<<<BB, CC, 0, stream>>>(w, Wg, Wb, Wls, bls, gamma, beta, lsv);
  ln_kernel<<<BB * NN, CC, 0, stream>>>(x, gamma, beta, hh, sq);

  pack_w_kernel<<<(128 * 64  + 255) / 256, 256, 0, stream>>>(W1,  wp,          128, 64);
  pack_w_kernel<<<(64  * 128 + 255) / 256, 256, 0, stream>>>(W2,  wp + 8192,   64,  128);
  pack_w_kernel<<<(256 * 128 + 255) / 256, 256, 0, stream>>>(Wx1, wp + 16384,  256, 128);
  pack_w_kernel<<<(128 * 128 + 255) / 256, 256, 0, stream>>>(Wx2, wp + 49152,  128, 128);

  knn_kernel<<<dim3(NN / 16, BB), 256, 166400, stream>>>(hh, sq, idx);

  edge_kernel<<<(BB * NN) / 8, 256, 163840, stream>>>(hh, idx, wp,
                                                      b1, b2, bx1, bx2,
                                                      lsv, x, out);
}
